// LSTMModel_54528904790837
// MI455X (gfx1250) — matrix operand order repaired and checked by reading, compiled, not run on hardware
//
#include <hip/hip_runtime.h>
#include <hip/hip_bf16.h>

#define H        128
#define T_ENC    256
#define T_DEC    32
#define BT       32            // batch rows per block
#define NTHREADS 512           // 16 waves (wave32)

typedef _Float16 v16h __attribute__((ext_vector_type(16)));
typedef _Float16 v8h  __attribute__((ext_vector_type(8)));
typedef float    v8f  __attribute__((ext_vector_type(8)));

// ---------------- LDS layout (bytes) ----------------
//   wih  [512][128] f16 : W_ih rows              @      0, 131072 B
//   whh  [512][128] f16 : W_hh rows              @ 131072, 131072 B
//   xb0  [ 32][128] f32 : x tile buffer 0        @ 262144,  16384 B
//   xb1  [ 32][128] f32 : x tile buffer 1        @ 278528,  16384 B
//   hbuf [ 32][128] f16 : h state                @ 294912,   8192 B
//   (decode: fc_W f16 [128][128] overlays wih)
#define OFF_WIH   0
#define OFF_WHH   131072
#define OFF_XB0   262144
#define OFF_XB1   278528
#define OFF_HBUF  294912
#define LDS_BYTES 303104

__device__ __forceinline__ float fast_sigmoid(float x) {
    return 1.0f / (1.0f + __expf(-x));
}
__device__ __forceinline__ float fast_tanh(float x) {
    float e = __expf(2.0f * x);
    return (e - 1.0f) / (e + 1.0f);
}

// ---- gfx1250 async copy: global -> LDS (ASYNCcnt) ----
__device__ __forceinline__ void async_g2l_b128(const float* __restrict__ g, void* lds_dst) {
    unsigned off = (unsigned)(size_t)lds_dst;   // low 32 bits of generic LDS ptr = LDS byte offset
    asm volatile("global_load_async_to_lds_b128 %0, %1, off"
                 :: "v"(off), "v"(g) : "memory");
}
__device__ __forceinline__ void wait_async0() {
    asm volatile("s_wait_asynccnt 0x0" ::: "memory");
}

// A-fragment (16x32 f16): lane covers M = lane%16,
// K = {half*16..+15} within the 32-wide k-tile.
__device__ __forceinline__ v16h a_frag_f16(const _Float16* row, int kofs, int half) {
    const _Float16* p0 = row + kofs + half * 16;
    v8h lo = *(const v8h*)p0;
    v8h hi = *(const v8h*)(p0 + 8);
    v16h f;
#pragma unroll
    for (int i = 0; i < 8; ++i) { f[i] = lo[i]; f[8 + i] = hi[i]; }
    return f;
}
// Same fragment but sourced from f32 LDS with in-register convert.
__device__ __forceinline__ v16h a_frag_f32(const float* row, int kofs, int half) {
    const float* p0 = row + kofs + half * 16;
    v8f lo = *(const v8f*)p0;
    v8f hi = *(const v8f*)(p0 + 8);
    v16h f;
#pragma unroll
    for (int i = 0; i < 8; ++i) { f[i] = (_Float16)lo[i]; f[8 + i] = (_Float16)hi[i]; }
    return f;
}

__global__ __launch_bounds__(NTHREADS)
void lstm_encdec_wmma(const float* __restrict__ x,
                      const float* __restrict__ W_ih,
                      const float* __restrict__ W_hh,
                      const float* __restrict__ b_ih,
                      const float* __restrict__ b_hh,
                      const float* __restrict__ fc_W,
                      const float* __restrict__ fc_b,
                      float* __restrict__ out) {
    extern __shared__ char smem[];
    _Float16* wih  = (_Float16*)(smem + OFF_WIH);
    _Float16* whh  = (_Float16*)(smem + OFF_WHH);
    _Float16* hbuf = (_Float16*)(smem + OFF_HBUF);
    _Float16* fcw  = wih;   // decode-phase overlay

    const int tid   = threadIdx.x;
    const int wave  = tid >> 5;
    const int lane  = tid & 31;
    const int lm    = lane & 15;      // N (B/C/D) / M (A)
    const int half  = lane >> 4;
    const int mtile = wave >> 3;      // 0..1
    const int jtile = wave & 7;       // 0..7
    const int b0    = blockIdx.x * BT;
    const int arow  = mtile * 16 + lm;

    // ---- one-time staging: weights f32->f16, h0 = 0 ----
    for (int idx = tid; idx < 512 * H; idx += NTHREADS) {
        wih[idx] = (_Float16)W_ih[idx];
        whh[idx] = (_Float16)W_hh[idx];
    }
    for (int idx = tid; idx < BT * H; idx += NTHREADS)
        hbuf[idx] = (_Float16)0.0f;

    // per-lane bias column values (added in the activation epilogue)
    float bias_g[4];
#pragma unroll
    for (int g = 0; g < 4; ++g) {
        int col = g * H + jtile * 16 + lm;
        bias_g[g] = b_ih[col] + b_hh[col];
    }
    const float fb = fc_b[jtile * 16 + lm];

    v8f creg = {};   // persistent cell tile (registers only)

    // ---- prime: async-copy x_0 into xb0 (2 x b128 per thread) ----
#pragma unroll
    for (int i = 0; i < 2; ++i) {
        int q = tid + i * NTHREADS;          // float4 index 0..1023
        int r = q >> 5, c = (q & 31) * 4;
        async_g2l_b128(&x[((size_t)(b0 + r) * T_ENC + 0) * (size_t)H + c],
                       (float*)(smem + OFF_XB0) + r * H + c);
    }

    // ======================= encode: 256 steps =======================
    // unroll-by-2 so the ping/pong parity is a compile-time constant and
    // all x-buffer accesses stay in the LDS (ds_load) address space.
#pragma unroll 2
    for (int t = 0; t < T_ENC; ++t) {
        const float* xcur = (const float*)(smem + ((t & 1) ? OFF_XB1 : OFF_XB0));
        float*       xnxt = (float*)      (smem + ((t & 1) ? OFF_XB0 : OFF_XB1));

        wait_async0();       // my async writes for step t done
        __syncthreads();     // x_t tile + h_t visible to all waves

        if (t + 1 < T_ENC) {  // overlap next tile fetch with this step's WMMAs
#pragma unroll
            for (int i = 0; i < 2; ++i) {
                int q = tid + i * NTHREADS;
                int r = q >> 5, c = (q & 31) * 4;
                async_g2l_b128(&x[((size_t)(b0 + r) * T_ENC + (t + 1)) * (size_t)H + c],
                               xnxt + r * H + c);
            }
        }
        if (t + 2 < T_ENC)   // pull t+2 toward L2 so the async engine hits cache
            __builtin_prefetch(
                &x[((size_t)(b0 + (tid >> 4)) * T_ENC + (t + 2)) * (size_t)H + (tid & 15) * 8],
                0, 0);

        // gates = x_t @ W_ih^T + h_t @ W_hh^T   (bias added in epilogue)
        v8f acc[4];
#pragma unroll
        for (int g = 0; g < 4; ++g)
#pragma unroll
            for (int r = 0; r < 8; ++r) acc[g][r] = 0.0f;

#pragma unroll
        for (int kt = 0; kt < 8; ++kt) {
            const bool xpart = (kt < 4);
            v16h af = xpart ? a_frag_f32(xcur + arow * H, kt * 32, half)
                            : a_frag_f16(hbuf + arow * H, (kt - 4) * 32, half);
            const _Float16* wsrc = xpart ? wih : whh;
            v16h bf[4];
#pragma unroll
            for (int g = 0; g < 4; ++g) {
                int n = g * H + jtile * 16 + lm;
                bf[g] = *(const v16h*)(wsrc + n * H + (kt & 3) * 32 + half * 16);
            }
#pragma unroll
            for (int g = 0; g < 4; ++g)
                acc[g] = __builtin_amdgcn_wmma_f32_16x16x32_f16(
                    false, af, false, bf[g], (short)0, acc[g], false, false);
        }
        __syncthreads();   // all LDS reads of this step done

        // pointwise LSTM cell in registers; h_{t+1} back to LDS as f16
#pragma unroll
        for (int r = 0; r < 8; ++r) {
            float iv = fast_sigmoid(acc[0][r] + bias_g[0]);
            float fv = fast_sigmoid(acc[1][r] + bias_g[1]);
            float gv = fast_tanh   (acc[2][r] + bias_g[2]);
            float ov = fast_sigmoid(acc[3][r] + bias_g[3]);
            float cn = fv * creg[r] + iv * gv;
            creg[r]  = cn;
            float hn = ov * fast_tanh(cn);
            int row  = mtile * 16 + r + 8 * half;   // D layout: M = r + 8*half
            hbuf[row * H + jtile * 16 + lm] = (_Float16)hn;
        }
    }

    // ---- decode prep: overlay fc_W into the (now dead) wih region ----
    __syncthreads();   // final h visible; wih reads finished
    for (int idx = tid; idx < H * H; idx += NTHREADS)
        fcw[idx] = (_Float16)fc_W[idx];
    __syncthreads();

    // ======================= decode: 32 steps =======================
    for (int s = 0; s < T_DEC; ++s) {
        // gates = h @ W_hh^T (+bias in epilogue); K = 128 -> 4 k-tiles
        v8f acc[4];
#pragma unroll
        for (int g = 0; g < 4; ++g)
#pragma unroll
            for (int r = 0; r < 8; ++r) acc[g][r] = 0.0f;

#pragma unroll
        for (int kt = 0; kt < 4; ++kt) {
            v16h af = a_frag_f16(hbuf + arow * H, kt * 32, half);
            v16h bf[4];
#pragma unroll
            for (int g = 0; g < 4; ++g) {
                int n = g * H + jtile * 16 + lm;
                bf[g] = *(const v16h*)(whh + n * H + kt * 32 + half * 16);
            }
#pragma unroll
            for (int g = 0; g < 4; ++g)
                acc[g] = __builtin_amdgcn_wmma_f32_16x16x32_f16(
                    false, af, false, bf[g], (short)0, acc[g], false, false);
        }
        __syncthreads();   // gate reads of h_s done

#pragma unroll
        for (int r = 0; r < 8; ++r) {
            float iv = fast_sigmoid(acc[0][r] + bias_g[0]);
            float fv = fast_sigmoid(acc[1][r] + bias_g[1]);
            float gv = fast_tanh   (acc[2][r] + bias_g[2]);
            float ov = fast_sigmoid(acc[3][r] + bias_g[3]);
            float cn = fv * creg[r] + iv * gv;
            creg[r]  = cn;
            float hn = ov * fast_tanh(cn);
            int row  = mtile * 16 + r + 8 * half;
            hbuf[row * H + jtile * 16 + lm] = (_Float16)hn;
        }
        __syncthreads();   // h_{s+1} visible

        // out = h_{s+1} @ fc_W^T + fc_b
        v8f oacc;
#pragma unroll
        for (int r = 0; r < 8; ++r) oacc[r] = fb;
#pragma unroll
        for (int kt = 0; kt < 4; ++kt) {
            v16h af = a_frag_f16(hbuf + arow * H, kt * 32, half);
            int n = jtile * 16 + lm;
            v16h bf = *(const v16h*)(fcw + n * H + kt * 32 + half * 16);
            oacc = __builtin_amdgcn_wmma_f32_16x16x32_f16(
                false, af, false, bf, (short)0, oacc, false, false);
        }
#pragma unroll
        for (int r = 0; r < 8; ++r) {
            int row = mtile * 16 + r + 8 * half;
            out[((size_t)(b0 + row) * T_DEC + s) * (size_t)H + jtile * 16 + lm] = oacc[r];
        }
    }
}

extern "C" void kernel_launch(void* const* d_in, const int* in_sizes, int n_in,
                              void* d_out, int out_size, void* d_ws, size_t ws_size,
                              hipStream_t stream) {
    (void)n_in; (void)d_ws; (void)ws_size; (void)out_size;
    const float* x    = (const float*)d_in[0];
    const float* W_ih = (const float*)d_in[1];
    const float* W_hh = (const float*)d_in[2];
    const float* b_ih = (const float*)d_in[3];
    const float* b_hh = (const float*)d_in[4];
    const float* fc_W = (const float*)d_in[5];
    const float* fc_b = (const float*)d_in[6];
    float* out = (float*)d_out;

    const int B = in_sizes[0] / (T_ENC * H);   // 4096
    const int nblocks = B / BT;                // 128

    static bool attr_set = false;
    if (!attr_set) {
        hipFuncSetAttribute((const void*)lstm_encdec_wmma,
                            hipFuncAttributeMaxDynamicSharedMemorySize, LDS_BYTES);
        attr_set = true;
    }

    lstm_encdec_wmma<<<nblocks, NTHREADS, LDS_BYTES, stream>>>(
        x, W_ih, W_hh, b_ih, b_hh, fc_W, fc_b, out);
}
